// KAGNN_8512625180982
// MI455X (gfx1250) — matrix operand order repaired and checked by reading, not compiled, not run on hardware
//
#include <hip/hip_runtime.h>
#include <hip/hip_bf16.h>
#include <stdint.h>

// ---------------------------------------------------------------------------
// KAGNN (Fourier-KAN GNN) for MI455X / gfx1250.
// KAN / linear GEMMs on v_wmma_f32_16x16x32_bf16. Fourier features are
// pre-expanded (sincos once + harmonic recurrences) into LDS as bf16 pairs in
// WMMA A-operand order, so the GEMM loop is pure ds_load_b128 + b128 + wmma.
// ---------------------------------------------------------------------------

#define DEV __device__ __forceinline__

typedef __attribute__((ext_vector_type(16))) __bf16 v16bf;
typedef __attribute__((ext_vector_type(8)))  float  v8f;

static constexpr int NN = 50000;   // nodes
static constexpr int EE = 200000;  // edges
static constexpr int GG = 64;      // graphs

// Packed-A row strides in dwords (3.5F dwords rounded up to x%4==0, and
// stride%64 == 36 -> distinct LDS banks across the 16 rows).
static constexpr int FTS_MSG = 676;  // F=192 -> 672 dwords used
static constexpr int FTS_UPD = 228;  // F=64  -> 224 dwords used

// RNE float -> bf16 bits
DEV unsigned short f2bf(float f) {
  union { float f; unsigned int u; } v; v.f = f;
  unsigned int u = v.u + 0x7FFFu + ((v.u >> 16) & 1u);
  return (unsigned short)(u >> 16);
}
DEV unsigned int pack2(float a, float b) {
  return (unsigned int)f2bf(a) | ((unsigned int)f2bf(b) << 16);
}

// A-matrix (16x32 bf16) K index for VGPR slot i, sub-element j, lane half.
DEV int a_k(int i, int j, int half) {
  int base = (i < 4) ? (i * 2) : (16 + (i - 4) * 2);
  return base + half * 8 + j;
}

// Fourier 6-pack for one feature value x, reference order
// [cos(1x),cos(2x),cos(3x),sin(1x),sin(2x),sin(3x)] -> 3 bf16-pair dwords.
DEV void fourier6(float x, unsigned int* d) {
  float s1 = __sinf(x), c1 = __cosf(x);
  float s1s1 = s1 * s1;
  float s2 = 2.0f * s1 * c1;
  float c2 = 1.0f - 2.0f * s1s1;
  float s3 = s1 * (3.0f - 4.0f * s1s1);
  float c3 = c1 * (4.0f * c1 * c1 - 3.0f);
  d[0] = pack2(c1, c2);
  d[1] = pack2(c3, s1);
  d[2] = pack2(s2, s3);
}

DEV float silu_f(float v) { return v / (1.0f + __expf(-v)); }

// ---------------------------------------------------------------------------
// GEMM with pre-packed bf16 A rows in LDS (arow = this lane's row, uint4*).
// Chunk c: A = dwords [c*16+half*4 .. +3] and [c*16+8+half*4 .. +3].
// ---------------------------------------------------------------------------
template <int NT>
DEV void wave_gemm_ldsA(const uint4* arow, int nchunks,
                        const uint4* __restrict__ packB, int lane, v8f* acc) {
  const int half = lane >> 4;
  for (int c = 0; c < nchunks; ++c) {
    union { uint4 q[2]; v16bf v; } a;
    a.q[0] = arow[c * 4 + half];
    a.q[1] = arow[c * 4 + 2 + half];
#pragma unroll
    for (int t = 0; t < NT; ++t) {
      const uint4* bp = packB + (size_t)((c * NT + t) * 32 + lane) * 2;
      union { uint4 q[2]; v16bf v; } b;
      b.q[0] = bp[0];
      b.q[1] = bp[1];
      acc[t] = __builtin_amdgcn_wmma_f32_16x16x32_bf16(
          false, a.v, false, b.v, (short)0, acc[t], false, false);
    }
  }
}

// Linear GEMM (single 32-wide chunk), A built from raw LDS floats.
template <int NT>
DEV void wave_gemm_lin(const float* xrow, int F, const uint4* __restrict__ packB,
                       int lane, v8f* acc) {
  const int half = lane >> 4;
  v16bf a;
#pragma unroll
  for (int i = 0; i < 8; ++i) {
#pragma unroll
    for (int j = 0; j < 2; ++j) {
      int k = a_k(i, j, half);
      a[i * 2 + j] = (__bf16)((k < F) ? xrow[k] : 0.0f);
    }
  }
#pragma unroll
  for (int t = 0; t < NT; ++t) {
    const uint4* bp = packB + (size_t)(t * 32 + lane) * 2;
    union { uint4 q[2]; v16bf v; } b;
    b.q[0] = bp[0];
    b.q[1] = bp[1];
    acc[t] = __builtin_amdgcn_wmma_f32_16x16x32_bf16(
        false, a, false, b.v, (short)0, acc[t], false, false);
  }
}

// ---------------------------------------------------------------------------
// Weight repack: f32 [w (OxwK) ‖ dc (OxdcK)] -> bf16 B-tiles in lane order.
// ---------------------------------------------------------------------------
__global__ void pack_kernel(const float* __restrict__ w, int wK,
                            const float* __restrict__ dc, int dcK,
                            int chunks, int ntiles,
                            unsigned int* __restrict__ out) {
  int i = blockIdx.x * blockDim.x + threadIdx.x;
  int total = chunks * ntiles * 256;
  if (i >= total) return;
  int dw = i & 7;
  int lane = (i >> 3) & 31;
  int rest = i >> 8;
  int t = rest % ntiles;
  int c = rest / ntiles;
  int n = t * 16 + (lane & 15);
  int half = lane >> 4;
  int k0 = c * 32 + ((dw < 4) ? (half * 8 + dw * 2) : (16 + half * 8 + (dw - 4) * 2));
  unsigned int r = 0;
  for (int j = 0; j < 2; ++j) {
    int k = k0 + j;
    float v = 0.0f;
    if (k < wK) v = w[(size_t)n * wK + k];
    else if (dc != nullptr && k < wK + dcK) v = dc[(size_t)n * dcK + (k - wK)];
    r |= (unsigned int)f2bf(v) << (16 * j);
  }
  out[i] = r;
}

__global__ void zero_kernel(float* __restrict__ p, int n) {
  int i = blockIdx.x * blockDim.x + threadIdx.x;
  if (i < n) p[i] = 0.0f;
}

// ---------------------------------------------------------------------------
// Embedding: out = SiLU(LN(x @ W^T + b)). 4 waves/block, wave owns 16 rows.
// ---------------------------------------------------------------------------
__global__ __launch_bounds__(128) void embed_kernel(
    const float* __restrict__ x, int F, const uint4* __restrict__ packB,
    const float* __restrict__ bias, const float* __restrict__ g,
    const float* __restrict__ be, float* __restrict__ out, int ntile) {
  __shared__ float xt[4][16][32];
  __shared__ float ot[4][16][64];
  __shared__ float mu[4][16], rs[4][16];
  const int lane = threadIdx.x & 31, w = threadIdx.x >> 5;
  const int tile = blockIdx.x * 4 + w;
  const bool valid = tile < ntile;
  if (valid) {
    for (int i = lane; i < 16 * F; i += 32) {
      int m = i / F, c = i - m * F;
      xt[w][m][c] = x[(size_t)(tile * 16 + m) * F + c];
    }
  }
  __syncthreads();
  v8f z8 = {0.f, 0.f, 0.f, 0.f, 0.f, 0.f, 0.f, 0.f};
  v8f acc[4] = {z8, z8, z8, z8};
  if (valid)
    wave_gemm_lin<4>(&xt[w][lane & 15][0], F, packB, lane, acc);
  if (valid) {
    int n0 = lane & 15, half = lane >> 4;
#pragma unroll
    for (int t = 0; t < 4; ++t) {
      float bn = bias[t * 16 + n0];
#pragma unroll
      for (int r = 0; r < 8; ++r)
        ot[w][r + half * 8][n0 + t * 16] = acc[t][r] + bn;
    }
  }
  __syncthreads();
  if (valid && lane < 16) {
    float s = 0.f, s2 = 0.f;
    for (int c = 0; c < 64; ++c) { float v = ot[w][lane][c]; s += v; s2 += v * v; }
    float m_ = s * (1.0f / 64.0f);
    float var = s2 * (1.0f / 64.0f) - m_ * m_;
    mu[w][lane] = m_;
    rs[w][lane] = rsqrtf(var + 1e-5f);
  }
  __syncthreads();
  if (valid) {
    for (int i = lane; i < 1024; i += 32) {
      int m = i >> 6, c = i & 63;
      float v = (ot[w][m][c] - mu[w][m]) * rs[w][m] * g[c] + be[c];
      out[(size_t)(tile * 16 + m) * 64 + c] = silu_f(v);
    }
  }
}

// ---------------------------------------------------------------------------
// Edge message: m = LN(SiLU(KAN([h_dst, h_src, e]))); scatter-add to aggr[dst].
// F = 192, Kdim = 1344 -> 42 chunks. 2 waves/block.
// ---------------------------------------------------------------------------
__global__ __launch_bounds__(64) void msg_kernel(
    const float* __restrict__ h, const float* __restrict__ e,
    const int* __restrict__ srcI, const int* __restrict__ dstI,
    const uint4* __restrict__ packB, const float* __restrict__ bias,
    const float* __restrict__ lng, const float* __restrict__ lnb,
    float* __restrict__ aggr, int ntile) {
  __shared__ float xt[2][16][192];
  __shared__ __align__(16) unsigned int ft[2][16 * FTS_MSG];
  __shared__ float ot[2][16][64];
  __shared__ int idd[2][16], iss[2][16];
  __shared__ float mu[2][16], rs[2][16];
  const int lane = threadIdx.x & 31, w = threadIdx.x >> 5;
  const int tile = blockIdx.x * 2 + w;
  const bool valid = tile < ntile;
  if (valid && lane < 16) {
    int eid = tile * 16 + lane;
    idd[w][lane] = dstI[eid];
    iss[w][lane] = srcI[eid];
  }
  __syncthreads();
  if (valid) {
    for (int i = lane; i < 16 * 192; i += 32) {
      int m = i / 192, c = i - m * 192;
      float v;
      if (c < 64)       v = h[(size_t)idd[w][m] * 64 + c];
      else if (c < 128) v = h[(size_t)iss[w][m] * 64 + (c - 64)];
      else              v = e[(size_t)(tile * 16 + m) * 64 + (c - 128)];
      xt[w][m][c] = v;
    }
  }
  __syncthreads();
  // Fourier expansion -> packed bf16 A rows.
  if (valid) {
    for (int i = lane; i < 16 * 192; i += 32) {
      int m = i / 192, f = i - m * 192;
      unsigned int d[3];
      fourier6(xt[w][m][f], d);
      unsigned int* row = &ft[w][m * FTS_MSG];
      row[3 * f + 0] = d[0];
      row[3 * f + 1] = d[1];
      row[3 * f + 2] = d[2];
    }
    for (int i = lane; i < 16 * 96; i += 32) {  // dc (raw) region, pairs
      int m = i / 96, dp = i - m * 96;
      ft[w][m * FTS_MSG + 576 + dp] = pack2(xt[w][m][2 * dp], xt[w][m][2 * dp + 1]);
    }
  }
  __syncthreads();
  v8f z8 = {0.f, 0.f, 0.f, 0.f, 0.f, 0.f, 0.f, 0.f};
  v8f acc[4] = {z8, z8, z8, z8};
  if (valid)
    wave_gemm_ldsA<4>((const uint4*)&ft[w][(lane & 15) * FTS_MSG], 42, packB,
                      lane, acc);
  if (valid) {
    int n0 = lane & 15, half = lane >> 4;
#pragma unroll
    for (int t = 0; t < 4; ++t) {
      float bn = bias[t * 16 + n0];
#pragma unroll
      for (int r = 0; r < 8; ++r)
        ot[w][r + half * 8][n0 + t * 16] = silu_f(acc[t][r] + bn);
    }
  }
  __syncthreads();
  if (valid && lane < 16) {
    float s = 0.f, s2 = 0.f;
    for (int c = 0; c < 64; ++c) { float v = ot[w][lane][c]; s += v; s2 += v * v; }
    float m_ = s * (1.0f / 64.0f);
    float var = s2 * (1.0f / 64.0f) - m_ * m_;
    mu[w][lane] = m_;
    rs[w][lane] = rsqrtf(var + 1e-5f);
  }
  __syncthreads();
  if (valid) {
    for (int i = lane; i < 1024; i += 32) {
      int m = i >> 6, c = i & 63;
      float v = (ot[w][m][c] - mu[w][m]) * rs[w][m] * lng[c] + lnb[c];
      atomicAdd(&aggr[(size_t)idd[w][m] * 64 + c], v);
    }
  }
}

// ---------------------------------------------------------------------------
// Node update: u = h + aggr; h = LN2(LN1(u + KAN(u))). F = 64 -> 14 chunks.
// ---------------------------------------------------------------------------
__global__ __launch_bounds__(128) void upd_kernel(
    const float* __restrict__ h, const float* __restrict__ aggr,
    const uint4* __restrict__ packB, const float* __restrict__ bias,
    const float* __restrict__ g1, const float* __restrict__ b1,
    const float* __restrict__ g2, const float* __restrict__ b2,
    float* __restrict__ hout, int ntile) {
  __shared__ float ut[4][16][64];
  __shared__ __align__(16) unsigned int ft[4][16 * FTS_UPD];
  __shared__ float yt[4][16][64];
  __shared__ float mu[4][16], rs[4][16];
  const int lane = threadIdx.x & 31, w = threadIdx.x >> 5;
  const int tile = blockIdx.x * 4 + w;
  const bool valid = tile < ntile;
  if (valid) {
    for (int i = lane; i < 1024; i += 32) {
      int m = i >> 6, c = i & 63;
      size_t gi = (size_t)(tile * 16 + m) * 64 + c;
      ut[w][m][c] = h[gi] + aggr[gi];
    }
  }
  __syncthreads();
  if (valid) {
    for (int i = lane; i < 16 * 64; i += 32) {
      int m = i >> 6, f = i & 63;
      unsigned int d[3];
      fourier6(ut[w][m][f], d);
      unsigned int* row = &ft[w][m * FTS_UPD];
      row[3 * f + 0] = d[0];
      row[3 * f + 1] = d[1];
      row[3 * f + 2] = d[2];
    }
    for (int i = lane; i < 16 * 32; i += 32) {
      int m = i >> 5, dp = i & 31;
      ft[w][m * FTS_UPD + 192 + dp] = pack2(ut[w][m][2 * dp], ut[w][m][2 * dp + 1]);
    }
  }
  __syncthreads();
  v8f z8 = {0.f, 0.f, 0.f, 0.f, 0.f, 0.f, 0.f, 0.f};
  v8f acc[4] = {z8, z8, z8, z8};
  if (valid)
    wave_gemm_ldsA<4>((const uint4*)&ft[w][(lane & 15) * FTS_UPD], 14, packB,
                      lane, acc);
  if (valid) {
    int n0 = lane & 15, half = lane >> 4;
#pragma unroll
    for (int t = 0; t < 4; ++t) {
      float bn = bias[t * 16 + n0];
#pragma unroll
      for (int r = 0; r < 8; ++r) {
        int m = r + half * 8, n = n0 + t * 16;
        yt[w][m][n] = ut[w][m][n] + acc[t][r] + bn;
      }
    }
  }
  __syncthreads();
  // LN1
  if (valid && lane < 16) {
    float s = 0.f, s2 = 0.f;
    for (int c = 0; c < 64; ++c) { float v = yt[w][lane][c]; s += v; s2 += v * v; }
    float m_ = s * (1.0f / 64.0f);
    float var = s2 * (1.0f / 64.0f) - m_ * m_;
    mu[w][lane] = m_;
    rs[w][lane] = rsqrtf(var + 1e-5f);
  }
  __syncthreads();
  if (valid) {
    for (int i = lane; i < 1024; i += 32) {
      int m = i >> 6, c = i & 63;
      yt[w][m][c] = (yt[w][m][c] - mu[w][m]) * rs[w][m] * g1[c] + b1[c];
    }
  }
  __syncthreads();
  // LN2
  if (valid && lane < 16) {
    float s = 0.f, s2 = 0.f;
    for (int c = 0; c < 64; ++c) { float v = yt[w][lane][c]; s += v; s2 += v * v; }
    float m_ = s * (1.0f / 64.0f);
    float var = s2 * (1.0f / 64.0f) - m_ * m_;
    mu[w][lane] = m_;
    rs[w][lane] = rsqrtf(var + 1e-5f);
  }
  __syncthreads();
  if (valid) {
    for (int i = lane; i < 1024; i += 32) {
      int m = i >> 6, c = i & 63;
      float v = (yt[w][m][c] - mu[w][m]) * rs[w][m] * g2[c] + b2[c];
      hout[(size_t)(tile * 16 + m) * 64 + c] = v;
    }
  }
}

// ---------------------------------------------------------------------------
// Global mean pool
// ---------------------------------------------------------------------------
__global__ void pool_sum_kernel(const float* __restrict__ h,
                                const int* __restrict__ batch,
                                float* __restrict__ sums) {
  int i = blockIdx.x * blockDim.x + threadIdx.x;
  if (i < NN * 64) {
    int n = i >> 6, c = i & 63;
    atomicAdd(&sums[(size_t)batch[n] * 64 + c], h[i]);
  }
}
__global__ void pool_cnt_kernel(const int* __restrict__ batch,
                                float* __restrict__ cnt) {
  int n = blockIdx.x * blockDim.x + threadIdx.x;
  if (n < NN) atomicAdd(&cnt[batch[n]], 1.0f);
}
__global__ void pool_div_kernel(const float* __restrict__ sums,
                                const float* __restrict__ cnt,
                                float* __restrict__ gp) {
  int i = blockIdx.x * blockDim.x + threadIdx.x;
  if (i < GG * 64) gp[i] = sums[i] / fmaxf(cnt[i >> 6], 1.0f);
}

// ---------------------------------------------------------------------------
// Head: z = SiLU(LN(KAN(g))) (O=32), out = z @ lin_w^T + lin_b. One block.
// ---------------------------------------------------------------------------
__global__ __launch_bounds__(128) void head_kernel(
    const float* __restrict__ gp, const uint4* __restrict__ packB,
    const float* __restrict__ bias, const float* __restrict__ g,
    const float* __restrict__ b, const float* __restrict__ lw,
    const float* __restrict__ lb, float* __restrict__ out) {
  __shared__ float xt[4][16][64];
  __shared__ __align__(16) unsigned int ft[4][16 * FTS_UPD];
  __shared__ float zt[64][32];
  __shared__ float mu[4][16], rs[4][16];
  const int lane = threadIdx.x & 31, w = threadIdx.x >> 5;
  const int tile = w;  // 4 tiles of 16 graphs
  for (int i = lane; i < 1024; i += 32) {
    int m = i >> 6, c = i & 63;
    xt[w][m][c] = gp[(size_t)(tile * 16 + m) * 64 + c];
  }
  __syncthreads();
  for (int i = lane; i < 16 * 64; i += 32) {
    int m = i >> 6, f = i & 63;
    unsigned int d[3];
    fourier6(xt[w][m][f], d);
    unsigned int* row = &ft[w][m * FTS_UPD];
    row[3 * f + 0] = d[0];
    row[3 * f + 1] = d[1];
    row[3 * f + 2] = d[2];
  }
  for (int i = lane; i < 16 * 32; i += 32) {
    int m = i >> 5, dp = i & 31;
    ft[w][m * FTS_UPD + 192 + dp] = pack2(xt[w][m][2 * dp], xt[w][m][2 * dp + 1]);
  }
  __syncthreads();
  v8f z8 = {0.f, 0.f, 0.f, 0.f, 0.f, 0.f, 0.f, 0.f};
  v8f acc[2] = {z8, z8};
  wave_gemm_ldsA<2>((const uint4*)&ft[w][(lane & 15) * FTS_UPD], 14, packB,
                    lane, acc);
  {
    int n0 = lane & 15, half = lane >> 4;
#pragma unroll
    for (int t = 0; t < 2; ++t) {
      float bn = bias[t * 16 + n0];
#pragma unroll
      for (int r = 0; r < 8; ++r)
        zt[tile * 16 + r + half * 8][n0 + t * 16] = acc[t][r] + bn;
    }
  }
  __syncthreads();
  if (lane < 16) {
    int row = tile * 16 + lane;
    float s = 0.f, s2 = 0.f;
    for (int c = 0; c < 32; ++c) { float v = zt[row][c]; s += v; s2 += v * v; }
    float m_ = s * (1.0f / 32.0f);
    float var = s2 * (1.0f / 32.0f) - m_ * m_;
    mu[w][lane] = m_;
    rs[w][lane] = rsqrtf(var + 1e-5f);
  }
  __syncthreads();
  for (int i = lane; i < 512; i += 32) {
    int m = i >> 5, c = i & 31;
    int row = tile * 16 + m;
    float v = (zt[row][c] - mu[w][m]) * rs[w][m] * g[c] + b[c];
    zt[row][c] = silu_f(v);
  }
  __syncthreads();
  int tid = threadIdx.x;
  if (tid < 64) {
    float s = lb[0];
    for (int c = 0; c < 32; ++c) s += zt[tid][c] * lw[c];
    out[tid] = s;
  }
}

// ---------------------------------------------------------------------------
// Host launcher
// ---------------------------------------------------------------------------
static inline void launch_pack(const float* w, int wK, const float* dc, int dcK,
                               int chunks, int ntiles, unsigned int* out,
                               hipStream_t s) {
  int total = chunks * ntiles * 256;
  pack_kernel<<<(total + 255) / 256, 256, 0, s>>>(w, wK, dc, dcK, chunks,
                                                  ntiles, out);
}

extern "C" void kernel_launch(void* const* d_in, const int* in_sizes, int n_in,
                              void* d_out, int out_size, void* d_ws,
                              size_t ws_size, hipStream_t stream) {
  (void)in_sizes; (void)n_in; (void)out_size; (void)ws_size;
  const float* x         = (const float*)d_in[0];
  const int*   eidx      = (const int*)d_in[1];   // jax x64 disabled -> int32
  const float* edge_attr = (const float*)d_in[2];
  const int*   batch     = (const int*)d_in[3];
  const float* ne_w = (const float*)d_in[4];
  const float* ne_b = (const float*)d_in[5];
  const float* ne_g = (const float*)d_in[6];
  const float* ne_be = (const float*)d_in[7];
  const float* ee_w = (const float*)d_in[8];
  const float* ee_b = (const float*)d_in[9];
  const float* ee_g = (const float*)d_in[10];
  const float* ee_be = (const float*)d_in[11];
  const float* c_msg_w[2]  = {(const float*)d_in[12], (const float*)d_in[22]};
  const float* c_msg_b[2]  = {(const float*)d_in[13], (const float*)d_in[23]};
  const float* c_msg_dc[2] = {(const float*)d_in[14], (const float*)d_in[24]};
  const float* c_mn_g[2]   = {(const float*)d_in[15], (const float*)d_in[25]};
  const float* c_mn_b[2]   = {(const float*)d_in[16], (const float*)d_in[26]};
  const float* c_upd_w[2]  = {(const float*)d_in[17], (const float*)d_in[27]};
  const float* c_upd_b[2]  = {(const float*)d_in[18], (const float*)d_in[28]};
  const float* c_upd_dc[2] = {(const float*)d_in[19], (const float*)d_in[29]};
  const float* c_un_g[2]   = {(const float*)d_in[20], (const float*)d_in[30]};
  const float* c_un_b[2]   = {(const float*)d_in[21], (const float*)d_in[31]};
  const float* n_g[2] = {(const float*)d_in[32], (const float*)d_in[34]};
  const float* n_b[2] = {(const float*)d_in[33], (const float*)d_in[35]};
  const float* hk_w  = (const float*)d_in[36];
  const float* hk_b  = (const float*)d_in[37];
  const float* hk_dc = (const float*)d_in[38];
  const float* hl_g  = (const float*)d_in[39];
  const float* hl_b  = (const float*)d_in[40];
  const float* lin_w = (const float*)d_in[41];
  const float* lin_b = (const float*)d_in[42];
  float* out = (float*)d_out;

  const int* srcI = eidx;        // edge_index[0]
  const int* dstI = eidx + EE;   // edge_index[1]

  // ---- workspace carve-up (float units)
  float* ws = (float*)d_ws;
  float* h    = ws;                      // N*64
  float* e    = h + (size_t)NN * 64;     // E*64
  float* aggr = e + (size_t)EE * 64;     // N*64
  float* sums = aggr + (size_t)NN * 64;  // G*64
  float* cnt  = sums + (size_t)GG * 64;  // G
  float* gp   = cnt + GG;                // G*64
  unsigned int* pk = (unsigned int*)(gp + (size_t)GG * 64);  // 16B aligned
  unsigned int* pk_ne   = pk;               // 1*4*256
  unsigned int* pk_ee   = pk_ne + 1024;     // 1*4*256
  unsigned int* pk_msg0 = pk_ee + 1024;     // 42*4*256
  unsigned int* pk_upd0 = pk_msg0 + 43008;  // 14*4*256
  unsigned int* pk_msg1 = pk_upd0 + 14336;  // 42*4*256
  unsigned int* pk_upd1 = pk_msg1 + 43008;  // 14*4*256
  unsigned int* pk_head = pk_upd1 + 14336;  // 14*2*256

  // ---- pack weights (bf16, B-tile layout)
  launch_pack(ne_w, 32, nullptr, 0, 1, 4, pk_ne, stream);
  launch_pack(ee_w, 16, nullptr, 0, 1, 4, pk_ee, stream);
  launch_pack(c_msg_w[0], 6 * 192, c_msg_dc[0], 192, 42, 4, pk_msg0, stream);
  launch_pack(c_upd_w[0], 6 * 64,  c_upd_dc[0], 64,  14, 4, pk_upd0, stream);
  launch_pack(c_msg_w[1], 6 * 192, c_msg_dc[1], 192, 42, 4, pk_msg1, stream);
  launch_pack(c_upd_w[1], 6 * 64,  c_upd_dc[1], 64,  14, 4, pk_upd1, stream);
  launch_pack(hk_w, 6 * 64, hk_dc, 64, 14, 2, pk_head, stream);

  // ---- embeddings
  {
    int nt = NN / 16;  // 3125
    embed_kernel<<<(nt + 3) / 4, 128, 0, stream>>>((const float*)x, 32,
        (const uint4*)pk_ne, ne_b, ne_g, ne_be, h, nt);
    int et = EE / 16;  // 12500
    embed_kernel<<<(et + 3) / 4, 128, 0, stream>>>(edge_attr, 16,
        (const uint4*)pk_ee, ee_b, ee_g, ee_be, e, et);
  }

  // ---- GNN layers
  unsigned int* pk_msg[2] = {pk_msg0, pk_msg1};
  unsigned int* pk_upd[2] = {pk_upd0, pk_upd1};
  for (int li = 0; li < 2; ++li) {
    zero_kernel<<<(NN * 64 + 255) / 256, 256, 0, stream>>>(aggr, NN * 64);
    int et = EE / 16;  // 12500 tiles, 2 waves/block
    msg_kernel<<<(et + 1) / 2, 64, 0, stream>>>(h, e, srcI, dstI,
        (const uint4*)pk_msg[li], c_msg_b[li], c_mn_g[li], c_mn_b[li], aggr, et);
    int nt = NN / 16;
    upd_kernel<<<(nt + 3) / 4, 128, 0, stream>>>(h, aggr,
        (const uint4*)pk_upd[li], c_upd_b[li], c_un_g[li], c_un_b[li],
        n_g[li], n_b[li], h, nt);
  }

  // ---- pooling
  zero_kernel<<<(GG * 64 + GG + 255) / 256, 256, 0, stream>>>(sums,
                                                              GG * 64 + GG);
  pool_sum_kernel<<<(NN * 64 + 255) / 256, 256, 0, stream>>>(h, batch, sums);
  pool_cnt_kernel<<<(NN + 255) / 256, 256, 0, stream>>>(batch, cnt);
  pool_div_kernel<<<(GG * 64 + 255) / 256, 256, 0, stream>>>(sums, cnt, gp);

  // ---- head
  head_kernel<<<1, 128, 0, stream>>>(gp, (const uint4*)pk_head, hk_b, hl_g,
                                     hl_b, lin_w, lin_b, out);
}
